// TemporalGNN_9569187135761
// MI455X (gfx1250) — compile-verified, matrix operand order checked by reading, not run on hardware
//
#include <hip/hip_runtime.h>
#include <hip/hip_bf16.h>

typedef __attribute__((ext_vector_type(16))) _Float16 v16h;
typedef __attribute__((ext_vector_type(8)))  float    v8f;

#define B_    16
#define T_    30
#define N_    2048
#define E_    32768
#define FIN   8
#define HIDD  64
#define TEMP_ 128
#define NCLS_ 2
#define EPS_  1e-5f
#define CH    24      // frames per chunk; 480 = 20 * 24

// ---------------------------------------------------------------------------
// K1: per-frame: stable argsort(node_ids) via bitonic sort in LDS, degree +
// dinv, GCN layer-1 matmul (K=8, scalar), zero agg buffer.
// ---------------------------------------------------------------------------
__global__ void __launch_bounds__(256)
k1_sort_deg_mm1(const float* __restrict__ x, const int* __restrict__ ei,
                const float* __restrict__ ew, const int* __restrict__ nid,
                const float* __restrict__ W1, float* __restrict__ dinv,
                float* __restrict__ hw, float* __restrict__ agg, int frame0)
{
    __shared__ unsigned keys[N_];
    __shared__ float    degf[N_];
    __shared__ float    W1s[FIN * HIDD];
    const int f   = blockIdx.x;
    const int g   = frame0 + f;
    const int tid = threadIdx.x;

    const int* nidp = nid + (size_t)g * N_;
    for (int i = tid; i < N_; i += 256) {
        keys[i] = (((unsigned)nidp[i]) << 11) | (unsigned)i;  // stable composite key
        degf[i] = 1.0f;                                       // self-loop degree
    }
    for (int i = tid; i < FIN * HIDD; i += 256) W1s[i] = W1[i];
    __syncthreads();

    // bitonic sort ascending (keys unique)
    for (int k = 2; k <= N_; k <<= 1) {
        for (int j = k >> 1; j > 0; j >>= 1) {
            for (int i = tid; i < N_; i += 256) {
                int ixj = i ^ j;
                if (ixj > i) {
                    unsigned a = keys[i], b = keys[ixj];
                    bool up = ((i & k) == 0);
                    if ((a > b) == up) { keys[i] = b; keys[ixj] = a; }
                }
            }
            __syncthreads();
        }
    }

    // weighted in-degree
    const int*   dstp = ei + (size_t)g * 2 * E_ + E_;
    const float* ewp  = ew + (size_t)g * E_;
    for (int e = tid; e < E_; e += 256) atomicAdd(&degf[dstp[e]], ewp[e]);
    __syncthreads();

    float* dinvp = dinv + (size_t)f * N_;
    for (int i = tid; i < N_; i += 256) dinvp[i] = rsqrtf(degf[i]);

    // GCN1: hw[r][c] = sum_k x[perm[r]][k] * W1[k][c]
    const float* xp  = x + (size_t)g * N_ * FIN;
    float*       hwp = hw + (size_t)f * N_ * HIDD;
    for (int r = tid; r < N_; r += 256) {
        int pr = (int)(keys[r] & 2047u);
        float xv[FIN];
        #pragma unroll
        for (int k = 0; k < FIN; k++) xv[k] = xp[pr * FIN + k];
        for (int c = 0; c < HIDD; c++) {
            float s = 0.f;
            #pragma unroll
            for (int k = 0; k < FIN; k++) s += xv[k] * W1s[k * HIDD + c];
            hwp[r * HIDD + c] = s;
        }
    }

    // zero agg region for this frame (scatter target)
    float4* aggp = (float4*)(agg + (size_t)f * N_ * HIDD);
    for (int i = tid; i < N_ * HIDD / 4; i += 256) aggp[i] = make_float4(0, 0, 0, 0);
}

// ---------------------------------------------------------------------------
// K2: edge scatter: agg[dst] += dinv[src]*w*dinv[dst] * hw[src]  (f32 atomics)
// ---------------------------------------------------------------------------
__global__ void __launch_bounds__(256)
k2_scatter(const int* __restrict__ ei, const float* __restrict__ ew,
           const float* __restrict__ dinv, const float* __restrict__ hw,
           float* __restrict__ agg, int frame0)
{
    int gid = blockIdx.x * 256 + threadIdx.x;   // CH*E threads exactly
    int f = gid / E_;
    int e = gid - f * E_;
    if (f >= CH) return;
    int g = frame0 + f;
    int src = ei[(size_t)g * 2 * E_ + e];
    int dst = ei[(size_t)g * 2 * E_ + E_ + e];
    float w = ew[(size_t)g * E_ + e];
    const float* dv = dinv + (size_t)f * N_;
    float norm = dv[src] * w * dv[dst];
    const float4* hs = (const float4*)(hw + ((size_t)f * N_ + src) * HIDD);
    float*        ad = agg + ((size_t)f * N_ + dst) * HIDD;
    #pragma unroll
    for (int q = 0; q < HIDD / 4; q++) {
        float4 hv = hs[q];
        atomicAdd(&ad[4 * q + 0], norm * hv.x);
        atomicAdd(&ad[4 * q + 1], norm * hv.y);
        atomicAdd(&ad[4 * q + 2], norm * hv.z);
        atomicAdd(&ad[4 * q + 3], norm * hv.w);
    }
}

// ---------------------------------------------------------------------------
// K3: GCN2 via WMMA f16->f32. A = relu(BN1(agg + dinv^2*hw + b1)) built
// directly in the CDNA5 16-bit A-fragment layout; B = f16(W2) from LDS.
// D overwrites hw; agg rows are re-zeroed for the layer-2 scatter.
// ---------------------------------------------------------------------------
__global__ void __launch_bounds__(256)
k3_wmma_mm2(const float* __restrict__ W2, const float* __restrict__ b1,
            const float* __restrict__ g1, const float* __restrict__ be1,
            const float* __restrict__ m1, const float* __restrict__ v1,
            const float* __restrict__ dinv, float* __restrict__ hw,
            float* __restrict__ agg)
{
    __shared__ _Float16 W2s[HIDD * HIDD];
    __shared__ float    s1[HIDD], t1[HIDD];
    const int f   = blockIdx.x;
    const int tid = threadIdx.x;

    for (int i = tid; i < HIDD * HIDD; i += 256) W2s[i] = (_Float16)W2[i];
    if (tid < HIDD) {
        float sc = g1[tid] * rsqrtf(v1[tid] + EPS_);
        s1[tid] = sc;
        t1[tid] = (b1[tid] - m1[tid]) * sc + be1[tid];  // folds GCN bias + BN shift
    }
    __syncthreads();

    const int wave = tid >> 5, lane = tid & 31;
    const int lm = lane & 15;    // M (A/D) or N (B/D) index
    const int lh = lane >> 4;    // half-wave select

    // hoist B fragments: bf[col_tile][k_step]
    v16h bf[4][2];
    #pragma unroll
    for (int ct = 0; ct < 4; ct++) {
        #pragma unroll
        for (int s = 0; s < 2; s++) {
            #pragma unroll
            for (int e = 0; e < 16; e++) {
                int K = (e & 7) + ((e >> 3) << 4) + (lh << 3) + 32 * s;
                bf[ct][s][e] = W2s[K * HIDD + ct * 16 + lm];
            }
        }
    }

    const float* dv  = dinv + (size_t)f * N_;
    float*       hwp = hw + (size_t)f * N_ * HIDD;
    float*       agp = agg + (size_t)f * N_ * HIDD;

    for (int i = 0; i < 16; i++) {                 // 16 row-tiles per wave
        int row0 = (wave * 16 + i) * 16;
        int row  = row0 + lm;
        float dsq = dv[row];
        dsq *= dsq;
        v16h a0, a1;
        #pragma unroll
        for (int e = 0; e < 16; e++) {
            int K  = (e & 7) + ((e >> 3) << 4) + (lh << 3);
            int K2 = K + 32;
            float r0 = agp[row * HIDD + K]  + dsq * hwp[row * HIDD + K];
            float r1 = agp[row * HIDD + K2] + dsq * hwp[row * HIDD + K2];
            a0[e] = (_Float16)fmaxf(r0 * s1[K]  + t1[K],  0.f);
            a1[e] = (_Float16)fmaxf(r1 * s1[K2] + t1[K2], 0.f);
        }
        #pragma unroll
        for (int ct = 0; ct < 4; ct++) {
            v8f acc = {};
            acc = __builtin_amdgcn_wmma_f32_16x16x32_f16(false, a0, false, bf[ct][0],
                                                         (short)0, acc, false, false);
            acc = __builtin_amdgcn_wmma_f32_16x16x32_f16(false, a1, false, bf[ct][1],
                                                         (short)0, acc, false, false);
            #pragma unroll
            for (int v = 0; v < 8; v++) {
                int orow = row0 + v + lh * 8;      // C/D layout: M = v + 8*half
                hwp[orow * HIDD + ct * 16 + lm] = acc[v];
            }
        }
        // zero this tile's agg rows (consumed above) for the layer-2 scatter
        float4* z = (float4*)&agp[row * HIDD + lh * 32];
        #pragma unroll
        for (int q = 0; q < 8; q++) z[q] = make_float4(0, 0, 0, 0);
    }
}

// ---------------------------------------------------------------------------
// K5: frames[g] = mean_rows relu(BN2(agg + dinv^2*hw + b2))
// ---------------------------------------------------------------------------
__global__ void __launch_bounds__(256)
k5_reduce(const float* __restrict__ b2, const float* __restrict__ g2,
          const float* __restrict__ be2, const float* __restrict__ m2,
          const float* __restrict__ v2, const float* __restrict__ dinv,
          const float* __restrict__ hw, const float* __restrict__ agg,
          float* __restrict__ frames, int frame0)
{
    __shared__ float part[4][HIDD];
    const int f   = blockIdx.x;
    const int tid = threadIdx.x;
    const int c  = tid & 63;
    const int rg = tid >> 6;
    float sc = g2[c] * rsqrtf(v2[c] + EPS_);
    float sh = (b2[c] - m2[c]) * sc + be2[c];
    const float* dv  = dinv + (size_t)f * N_;
    const float* hwp = hw + (size_t)f * N_ * HIDD;
    const float* agp = agg + (size_t)f * N_ * HIDD;
    float sum = 0.f;
    for (int r = rg; r < N_; r += 4) {
        float d = dv[r];
        float raw = agp[r * HIDD + c] + d * d * hwp[r * HIDD + c];
        sum += fmaxf(raw * sc + sh, 0.f);
    }
    part[rg][c] = sum;
    __syncthreads();
    if (rg == 0)
        frames[(size_t)(frame0 + f) * HIDD + c] =
            (part[0][c] + part[1][c] + part[2][c] + part[3][c]) * (1.0f / N_);
}

// ---------------------------------------------------------------------------
// K6: GRU over T steps (frames batch-major, b*T + t) + 2-layer classifier. Single block.
// ---------------------------------------------------------------------------
__global__ void __launch_bounds__(256)
k6_gru(const float* __restrict__ frames, const float* __restrict__ Wih,
       const float* __restrict__ Whh, const float* __restrict__ bih,
       const float* __restrict__ bhh, const float* __restrict__ Wc1,
       const float* __restrict__ bc1, const float* __restrict__ Wc2,
       const float* __restrict__ bc2, float* __restrict__ out)
{
    __shared__ float h[B_][TEMP_];
    __shared__ float gi[B_][3 * TEMP_];
    __shared__ float gh[B_][3 * TEMP_];
    __shared__ float hid[B_][TEMP_ / 2];
    const int tid = threadIdx.x;
    for (int i = tid; i < B_ * TEMP_; i += 256) h[i / TEMP_][i % TEMP_] = 0.f;
    __syncthreads();

    for (int t = 0; t < T_; t++) {
        for (int idx = tid; idx < B_ * 3 * TEMP_; idx += 256) {
            int b = idx / (3 * TEMP_), j = idx % (3 * TEMP_);
            const float* xt = frames + (size_t)(b * T_ + t) * HIDD;
            float si = bih[j];
            for (int k = 0; k < HIDD; k++) si += xt[k] * Wih[j * HIDD + k];
            gi[b][j] = si;
            float sh = bhh[j];
            for (int k = 0; k < TEMP_; k++) sh += h[b][k] * Whh[j * TEMP_ + k];
            gh[b][j] = sh;
        }
        __syncthreads();
        for (int idx = tid; idx < B_ * TEMP_; idx += 256) {
            int b = idx / TEMP_, j = idx % TEMP_;
            float r = 1.f / (1.f + expf(-(gi[b][j] + gh[b][j])));
            float z = 1.f / (1.f + expf(-(gi[b][TEMP_ + j] + gh[b][TEMP_ + j])));
            float n = tanhf(gi[b][2 * TEMP_ + j] + r * gh[b][2 * TEMP_ + j]);
            h[b][j] = (1.f - z) * n + z * h[b][j];
        }
        __syncthreads();
    }
    for (int idx = tid; idx < B_ * (TEMP_ / 2); idx += 256) {
        int b = idx / (TEMP_ / 2), j = idx % (TEMP_ / 2);
        float s = bc1[j];
        for (int k = 0; k < TEMP_; k++) s += h[b][k] * Wc1[k * (TEMP_ / 2) + j];
        hid[b][j] = fmaxf(s, 0.f);
    }
    __syncthreads();
    if (tid < B_ * NCLS_) {
        int b = tid / NCLS_, cc = tid % NCLS_;
        float s = bc2[cc];
        for (int k = 0; k < TEMP_ / 2; k++) s += hid[b][k] * Wc2[k * NCLS_ + cc];
        out[b * NCLS_ + cc] = s;
    }
}

// ---------------------------------------------------------------------------
extern "C" void kernel_launch(void* const* d_in, const int* in_sizes, int n_in,
                              void* d_out, int out_size, void* d_ws, size_t ws_size,
                              hipStream_t stream)
{
    (void)in_sizes; (void)n_in; (void)out_size; (void)ws_size;
    const float* x   = (const float*)d_in[0];
    const int*   ei  = (const int*)d_in[1];
    const float* ew  = (const float*)d_in[2];
    const int*   nid = (const int*)d_in[3];
    const float* W1  = (const float*)d_in[4];
    const float* b1  = (const float*)d_in[5];
    const float* W2  = (const float*)d_in[6];
    const float* b2  = (const float*)d_in[7];
    const float* g1  = (const float*)d_in[8];
    const float* be1 = (const float*)d_in[9];
    const float* m1  = (const float*)d_in[10];
    const float* v1  = (const float*)d_in[11];
    const float* g2  = (const float*)d_in[12];
    const float* be2 = (const float*)d_in[13];
    const float* m2  = (const float*)d_in[14];
    const float* v2  = (const float*)d_in[15];
    const float* Wih = (const float*)d_in[16];
    const float* Whh = (const float*)d_in[17];
    const float* bih = (const float*)d_in[18];
    const float* bhh = (const float*)d_in[19];
    const float* Wc1 = (const float*)d_in[20];
    const float* bc1 = (const float*)d_in[21];
    const float* Wc2 = (const float*)d_in[22];
    const float* bc2 = (const float*)d_in[23];
    float* out = (float*)d_out;

    float* ws     = (float*)d_ws;
    float* dinv   = ws;                                  // CH*N
    float* hw     = dinv + (size_t)CH * N_;              // CH*N*HIDD
    float* agg    = hw + (size_t)CH * N_ * HIDD;         // CH*N*HIDD
    float* frames = agg + (size_t)CH * N_ * HIDD;        // B*T*HIDD

    const int nchunks = (B_ * T_) / CH;
    for (int chunk = 0; chunk < nchunks; chunk++) {
        int f0 = chunk * CH;
        k1_sort_deg_mm1<<<CH, 256, 0, stream>>>(x, ei, ew, nid, W1, dinv, hw, agg, f0);
        k2_scatter<<<CH * E_ / 256, 256, 0, stream>>>(ei, ew, dinv, hw, agg, f0);
        k3_wmma_mm2<<<CH, 256, 0, stream>>>(W2, b1, g1, be1, m1, v1, dinv, hw, agg);
        k2_scatter<<<CH * E_ / 256, 256, 0, stream>>>(ei, ew, dinv, hw, agg, f0);
        k5_reduce<<<CH, 256, 0, stream>>>(b2, g2, be2, m2, v2, dinv, hw, agg, frames, f0);
    }
    k6_gru<<<1, 256, 0, stream>>>(frames, Wih, Whh, bih, bhh, Wc1, bc1, Wc2, bc2, out);
}
